// MoEMLP_38422777430717
// MI455X (gfx1250) — matrix operand order repaired and checked by reading, not compiled, not run on hardware
//
#include <hip/hip_runtime.h>

typedef unsigned short u16;
typedef unsigned int   u32;

#define TOK   8192
#define DDIM  1024
#define HDIM  1024
#define NEXP  64
#define TOPK  6
#define CAPE  1280
#define NPAIR (TOK*TOPK)

typedef __attribute__((ext_vector_type(16))) __bf16 v16bf;
typedef __attribute__((ext_vector_type(8)))  float  v8f;

template <typename T, typename F>
__device__ __forceinline__ T bitcast(const F& f) {
  static_assert(sizeof(T) == sizeof(F), "size mismatch");
  T t; __builtin_memcpy(&t, &f, sizeof(T)); return t;
}

// f32 -> bf16, round-half-up (differs from RNE only on exact ties).
__device__ __forceinline__ u16 f2bf(float f) {
  return (u16)((bitcast<u32>(f) + 0x8000u) >> 16);
}
__device__ __forceinline__ float bf2f(u16 h) {
  u32 u = ((u32)h) << 16;
  return bitcast<float>(u);
}

// Pack two f32 -> bf16 pair: prefer v_cvt_pk_bf16_f32, else 2 adds + v_perm_b32.
__device__ __forceinline__ u32 pk_bf16(float lo, float hi) {
#if __has_builtin(__builtin_amdgcn_cvt_pk_bf16_f32)
  auto r = __builtin_amdgcn_cvt_pk_bf16_f32(lo, hi);
  return bitcast<u32>(r);
#else
  u32 a = bitcast<u32>(lo) + 0x8000u;
  u32 b = bitcast<u32>(hi) + 0x8000u;
  return __builtin_amdgcn_perm(b, a, 0x07060302u);  // {b[31:16], a[31:16]}
#endif
}

// gfx1250 async copy: global -> LDS, 16B per lane, tracked by ASYNCcnt.
// VDST holds the wave-relative LDS byte offset (= low 32 bits of the generic
// pointer to a __shared__ object, per the LDS aperture mapping).
__device__ __forceinline__ void async_g2l_b128(u32 ldsOff, const u16* gsrc) {
  asm volatile("global_load_async_to_lds_b128 %0, %1, off"
               :: "v"(ldsOff), "v"(gsrc) : "memory");
}
__device__ __forceinline__ void wait_asynccnt0() {
  asm volatile("s_wait_asynccnt 0" ::: "memory");
}

// ---------------------------------------------------------------- utilities
__global__ void init_kernel(int* counts) {
  if (threadIdx.x < NEXP) counts[threadIdx.x] = 0;
}

__global__ void cvt_f32_bf16(const float* __restrict__ src, u16* __restrict__ dst, long n4) {
  long i = (long)blockIdx.x * blockDim.x + threadIdx.x;
  if (i >= n4) return;
  float4 f = ((const float4*)src)[i];
  u32 h[2] = { pk_bf16(f.x, f.y), pk_bf16(f.z, f.w) };
  ((uint2*)dst)[i] = bitcast<uint2>(h);
}

// ---------------------------------------------------------------- gating
__global__ void gate_topk_kernel(const float* __restrict__ x, const float* __restrict__ Wg,
                                 int* __restrict__ topI, float* __restrict__ topS) {
  __shared__ float sl[NEXP];
  const int t = blockIdx.x, e = threadIdx.x;
  const float* xr = x + (size_t)t * DDIM;
  float acc = 0.f;
  for (int d = 0; d < DDIM; ++d) acc = fmaf(xr[d], Wg[d * NEXP + e], acc);
  sl[e] = acc;
  __syncthreads();
  if (e == 0) {
    float mx = sl[0];
    for (int i = 1; i < NEXP; ++i) mx = fmaxf(mx, sl[i]);
    float s = 0.f;
    for (int i = 0; i < NEXP; ++i) { sl[i] = expf(sl[i] - mx); s += sl[i]; }
    float inv = 1.f / s;
    for (int k = 0; k < TOPK; ++k) {
      int bi = 0; float bv = -1.f;
      for (int i = 0; i < NEXP; ++i) if (sl[i] > bv) { bv = sl[i]; bi = i; }
      topI[t * TOPK + k] = bi;
      topS[t * TOPK + k] = bv * inv;
      sl[bi] = -1.f;
    }
  }
}

__global__ void dispatch_kernel(const int* __restrict__ topI, int* counts,
                                int* __restrict__ slotTok, int* __restrict__ pairRank) {
  int p = blockIdx.x * blockDim.x + threadIdx.x;
  if (p >= NPAIR) return;
  int e = topI[p];
  int r = atomicAdd(&counts[e], 1);
  if (r < CAPE) { slotTok[e * CAPE + r] = p / TOPK; pairRank[p] = r; }
  else pairRank[p] = -1;
}

// ---------------------------------------------------------------- WMMA GEMM
// C[M,N] = act( A[M,K](bf16, optionally row-gathered) * B[K,N](f32->bf16) + bias )
// 256 threads = 8 waves; block tile 128x128, K-step 32; wave tile 64x32 (4x2 wmma).
// Double-buffered LDS; A tiles via GLOBAL_LOAD_ASYNC_TO_LDS_B128 (ASYNCcnt),
// B tiles via fp32 global loads staged in registers, converted after compute.
#define BM 128
#define BN 128
#define BK 32
#define LDT 40   // padded LDS row stride (u16 units) -> conflict-free b128 frags

template<bool GATHER, bool GELU, bool OUTBF16>
__global__ void __launch_bounds__(256)
gemm_bf16_wmma(const u16* __restrict__ A, const int* __restrict__ gatherTok,
               const int* __restrict__ counts,
               const float* __restrict__ B, const float* __restrict__ bias,
               void* __restrict__ Cout, int Kd, int Nn, int Mrows) {
  __shared__ u16 ldsA[2][BM * LDT];   // [buf][row][k]
  __shared__ u16 ldsB[2][BN * LDT];   // [buf][col][k] (transposed weights)

  const int e = blockIdx.z;
  const int rowBase = blockIdx.y * BM;
  const int colBase = blockIdx.x * BN;
  int cnt = Mrows;
  if (counts) { cnt = counts[e]; if (rowBase >= cnt) return; }

  const size_t bOff    = (size_t)e * Kd * Nn;
  const size_t biasOff = (size_t)e * Nn;
  const size_t cRowOff = (size_t)e * Mrows;

  const int tid  = threadIdx.x;
  const int lane = tid & 31;
  const int wid  = tid >> 5;
  const int wm   = (wid & 1) * 64;   // wave row offset
  const int wn   = (wid >> 1) * 32;  // wave col offset

  // A tile loader: thread -> (row 0..127, 16-elem half)
  const int ar = tid >> 1;
  const int ah = (tid & 1) * 16;
  const u16* aSrc;
  {
    int gr = rowBase + ar;
    if (GATHER) {
      int tok = (gr < cnt) ? gatherTok[e * CAPE + gr] : 0;
      aSrc = A + (size_t)tok * Kd + ah;
    } else {
      aSrc = A + ((size_t)e * Mrows + gr) * Kd + ah;
    }
  }
  // B tile loader: thread -> (col 0..127, 16-row half)
  const int bcol = tid & 127;
  const int bkh  = (tid >> 7) * 16;

  // Precomputed wave-relative LDS byte offsets for the async A copies.
  u32 aLds[2];
  aLds[0] = (u32)(size_t)&ldsA[0][ar * LDT + ah];
  aLds[1] = (u32)(size_t)&ldsA[1][ar * LDT + ah];

  v8f acc[4][2];
  const v8f vz = {0.f,0.f,0.f,0.f,0.f,0.f,0.f,0.f};
  for (int i = 0; i < 4; ++i) for (int j = 0; j < 2; ++j) acc[i][j] = vz;

  const int fr   = lane & 15;
  const int aklo = (lane >> 4) << 3;    // A frag: 0 or 8  (K 0-7/16-23 vs 8-15/24-31)
  const int bklo = (lane >> 4) << 3;    // B frag: 0 or 8  (K 0-7/16-23 vs 8-15/24-31)

  float fb[16];   // staged fp32 weight tile values

  auto issueA = [&](int k0, int buf) {
    const u16* s = aSrc + k0;
    async_g2l_b128(aLds[buf],      s);
    async_g2l_b128(aLds[buf] + 16, s + 8);
  };
  auto loadB = [&](int k0) {
    const float* bp = B + bOff + (size_t)(k0 + bkh) * Nn + colBase + bcol;
    #pragma unroll
    for (int kk = 0; kk < 16; ++kk) fb[kk] = bp[kk * Nn];
  };
  auto prefetchB = [&](int k0) {   // warm GL2 two K-steps ahead
    const float* bp = B + bOff + (size_t)(k0 + bkh) * Nn + colBase + bcol;
    __builtin_prefetch(bp, 0, 1);
    __builtin_prefetch(bp + 8 * (size_t)Nn, 0, 1);
  };
  auto storeB = [&](int buf) {
    u32 pk[8];
    #pragma unroll
    for (int kk = 0; kk < 8; ++kk)
      pk[kk] = pk_bf16(fb[2 * kk], fb[2 * kk + 1]);
    uint4* d = (uint4*)&ldsB[buf][bcol * LDT + bkh];
    d[0] = make_uint4(pk[0], pk[1], pk[2], pk[3]);
    d[1] = make_uint4(pk[4], pk[5], pk[6], pk[7]);
  };

  const int NK = Kd / BK;

  // prologue: fully stage tile 0 into buffer 0
  issueA(0, 0);
  loadB(0);
  if (NK > 1) prefetchB(BK);
  storeB(0);

  for (int i = 0; i < NK; ++i) {
    const int cur = i & 1;
    const int nxt = cur ^ 1;
    wait_asynccnt0();        // our async A copies into `cur` are complete
    __syncthreads();         // all waves' LDS writes for tile i visible
    const bool more = (i + 1 < NK);
    if (more) {              // kick off tile i+1 while we compute tile i
      issueA((i + 1) * BK, nxt);
      loadB((i + 1) * BK);
      if (i + 2 < NK) prefetchB((i + 2) * BK);
    }

    // fragments + 8 WMMA on tile i
    v16bf af[4], bfr[2];
    #pragma unroll
    for (int ii = 0; ii < 4; ++ii) {
      const u16* p = &ldsA[cur][(wm + ii * 16 + fr) * LDT];
      struct { uint4 a, b; } f;
      f.a = *(const uint4*)(p + aklo);
      f.b = *(const uint4*)(p + 16 + aklo);
      af[ii] = bitcast<v16bf>(f);
    }
    #pragma unroll
    for (int j = 0; j < 2; ++j) {
      const u16* p = &ldsB[cur][(wn + j * 16 + fr) * LDT + bklo];
      struct { uint4 a, b; } f;
      f.a = *(const uint4*)(p);
      f.b = *(const uint4*)(p + 16);
      bfr[j] = bitcast<v16bf>(f);
    }
    #pragma unroll
    for (int ii = 0; ii < 4; ++ii)
      #pragma unroll
      for (int j = 0; j < 2; ++j)
        acc[ii][j] = __builtin_amdgcn_wmma_f32_16x16x32_bf16(
            false, af[ii], false, bfr[j], (short)0, acc[ii][j], false, false);

    if (more) storeB(nxt);   // convert staged weights after compute is issued
  }

  // epilogue: C/D layout -> lane n = lane&15, VGPR r holds M = r (+8 if lane>=16)
  const int rhi = (lane >> 4) << 3;
  #pragma unroll
  for (int j = 0; j < 2; ++j) {
    int col = colBase + wn + j * 16 + fr;
    float bv = bias[biasOff + col];
    #pragma unroll
    for (int i = 0; i < 4; ++i) {
      #pragma unroll
      for (int r = 0; r < 8; ++r) {
        int row = rowBase + wm + i * 16 + rhi + r;
        float v = acc[i][j][r] + bv;
        if (GELU) v = 0.5f * v * (1.f + erff(v * 0.70710678118654752f));
        size_t idx = (cRowOff + row) * (size_t)Nn + col;
        if (OUTBF16) ((u16*)Cout)[idx] = f2bf(v);
        else         ((float*)Cout)[idx] = v;
      }
    }
  }
}

// ---------------------------------------------------------------- combine
__global__ void combine_kernel(const int* __restrict__ topI, const float* __restrict__ topS,
                               const int* __restrict__ pairRank,
                               const u16* __restrict__ ybuf, float* __restrict__ out) {
  int t = blockIdx.x;
  int d0 = threadIdx.x * 4;
  float4 o = *(float4*)&out[(size_t)t * DDIM + d0];
  for (int k = 0; k < TOPK; ++k) {
    int p = t * TOPK + k;
    int r = pairRank[p];
    if (r < 0) continue;
    int e = topI[p];
    float sc = topS[p];
    const u16* yr = ybuf + ((size_t)e * CAPE + r) * DDIM + d0;
    uint2 q = *(const uint2*)yr;
    o.x += sc * bf2f((u16)(q.x & 0xffff));
    o.y += sc * bf2f((u16)(q.x >> 16));
    o.z += sc * bf2f((u16)(q.y & 0xffff));
    o.w += sc * bf2f((u16)(q.y >> 16));
  }
  *(float4*)&out[(size_t)t * DDIM + d0] = o;
}

// ---------------------------------------------------------------- stats
__global__ void stats_kernel(const int* __restrict__ topI, const float* __restrict__ topS,
                             const int* __restrict__ counts, float* __restrict__ out) {
  __shared__ float red[256];
  int e = blockIdx.x;
  float s = 0.f;
  for (int p = threadIdx.x; p < NPAIR; p += 256)
    if (topI[p] == e) s += topS[p];
  red[threadIdx.x] = s;
  __syncthreads();
  for (int w = 128; w > 0; w >>= 1) {
    if (threadIdx.x < w) red[threadIdx.x] += red[threadIdx.x + w];
    __syncthreads();
  }
  if (threadIdx.x == 0) {
    const float invP = 1.f / (float)NPAIR;
    out[(size_t)TOK * DDIM + 1 + e]        = (float)counts[e] * invP;  // expert_freq
    out[(size_t)TOK * DDIM + 1 + NEXP + e] = red[0] * invP;            // expert_prob
  }
}

__global__ void balance_kernel(float* __restrict__ out) {
  __shared__ float red[NEXP];
  int e = threadIdx.x;
  float f = out[(size_t)TOK * DDIM + 1 + e];
  float p = out[(size_t)TOK * DDIM + 1 + NEXP + e];
  red[e] = f * p;
  __syncthreads();
  if (e == 0) {
    float s = 0.f;
    for (int i = 0; i < NEXP; ++i) s += red[i];
    out[(size_t)TOK * DDIM] = 0.01f * s;   // balance_loss
  }
}

// ---------------------------------------------------------------- launcher
extern "C" void kernel_launch(void* const* d_in, const int* in_sizes, int n_in,
                              void* d_out, int out_size, void* d_ws, size_t ws_size,
                              hipStream_t stream) {
  const float* x   = (const float*)d_in[0];
  const float* Wg  = (const float*)d_in[1];
  const float* Ws1 = (const float*)d_in[2];
  const float* bs1 = (const float*)d_in[3];
  const float* Ws2 = (const float*)d_in[4];
  const float* bs2 = (const float*)d_in[5];
  const float* W1  = (const float*)d_in[6];
  const float* b1  = (const float*)d_in[7];
  const float* W2  = (const float*)d_in[8];
  const float* b2  = (const float*)d_in[9];
  float* out = (float*)d_out;
  (void)in_sizes; (void)n_in; (void)out_size; (void)ws_size;

  char* ws = (char*)d_ws;
  size_t off = 0;
  auto take = [&](size_t bytes) -> char* {
    char* p = ws + off;
    off = (off + bytes + 255) & ~(size_t)255;
    return p;
  };
  u16*   xb       = (u16*)take((size_t)TOK * DDIM * 2);
  u16*   hsb      = (u16*)take((size_t)TOK * HDIM * 2);
  u16*   hbuf     = (u16*)take((size_t)NEXP * CAPE * HDIM * 2);
  u16*   ybuf     = (u16*)take((size_t)NEXP * CAPE * DDIM * 2);
  int*   topI     = (int*)take((size_t)NPAIR * 4);
  float* topS     = (float*)take((size_t)NPAIR * 4);
  int*   pairRank = (int*)take((size_t)NPAIR * 4);
  int*   slotTok  = (int*)take((size_t)NEXP * CAPE * 4);
  int*   counts   = (int*)take(256);

  init_kernel<<<dim3(1), dim3(64), 0, stream>>>(counts);

  long n4 = (long)TOK * DDIM / 4;
  cvt_f32_bf16<<<dim3((unsigned)((n4 + 255) / 256)), dim3(256), 0, stream>>>(x, xb, n4);

  gate_topk_kernel<<<dim3(TOK), dim3(NEXP), 0, stream>>>(x, Wg, topI, topS);
  dispatch_kernel<<<dim3(NPAIR / 256), dim3(256), 0, stream>>>(topI, counts, slotTok, pairRank);

  // shared expert: out = gelu(xb@Ws1+bs1)@Ws2+bs2
  gemm_bf16_wmma<false, true, true><<<dim3(HDIM / 128, TOK / 128, 1), dim3(256), 0, stream>>>(
      xb, nullptr, nullptr, Ws1, bs1, hsb, DDIM, HDIM, TOK);
  gemm_bf16_wmma<false, false, false><<<dim3(DDIM / 128, TOK / 128, 1), dim3(256), 0, stream>>>(
      hsb, nullptr, nullptr, Ws2, bs2, out, HDIM, DDIM, TOK);

  // routed experts: gather-A GEMM -> gelu -> hbuf; hbuf -> ybuf
  gemm_bf16_wmma<true, true, true><<<dim3(HDIM / 128, CAPE / 128, NEXP), dim3(256), 0, stream>>>(
      xb, slotTok, counts, W1, b1, hbuf, DDIM, HDIM, CAPE);
  gemm_bf16_wmma<false, false, true><<<dim3(DDIM / 128, CAPE / 128, NEXP), dim3(256), 0, stream>>>(
      hbuf, nullptr, counts, W2, b2, ybuf, HDIM, DDIM, CAPE);

  combine_kernel<<<dim3(TOK), dim3(256), 0, stream>>>(topI, topS, pairRank, ybuf, out);
  stats_kernel<<<dim3(NEXP), dim3(256), 0, stream>>>(topI, topS, counts, out);
  balance_kernel<<<dim3(1), dim3(NEXP), 0, stream>>>(out);
}
